// TorchCausalSelfAttention_730144441055
// MI455X (gfx1250) — matrix operand order repaired and checked by reading, not compiled, not run on hardware
//
#include <hip/hip_runtime.h>
#include <hip/hip_bf16.h>

typedef unsigned short u16;
typedef unsigned int   u32;

typedef __attribute__((ext_vector_type(16))) __bf16 v16bf;
typedef __attribute__((ext_vector_type(8)))  float  v8f;

union Frag {
    v16bf v;
    uint4 q[2];
};

static constexpr int B_ = 2;
static constexpr int T_ = 2048;
static constexpr int C_ = 1024;
static constexpr int H_ = 16;
static constexpr int D_ = 64;
static constexpr float LOG2E = 1.44269504088896340736f;
static constexpr float NEG_BIG = -1e30f;

__device__ __forceinline__ u16 f2bf(float x) {
    u32 u = __float_as_uint(x);
    u32 r = u + 0x7FFFu + ((u >> 16) & 1u);   // round-to-nearest-even
    return (u16)(r >> 16);
}

#define WMMA_BF16(a, b, c) \
    __builtin_amdgcn_wmma_f32_16x16x32_bf16(false, (a), false, (b), (short)0, (c), false, false)

// ---------------------------------------------------------------------------
// Kernel 0: fp32 -> bf16 conversion of x and the four weight matrices
// ---------------------------------------------------------------------------
__global__ __launch_bounds__(256) void cvt_bf16_kernel(
    const float* __restrict__ x,
    const float* __restrict__ wq, const float* __restrict__ wk,
    const float* __restrict__ wv, const float* __restrict__ wp,
    u16* __restrict__ xb, u16* __restrict__ wb)
{
    const size_t NX = (size_t)B_ * T_ * C_;      // 4,194,304
    const size_t NW = (size_t)C_ * C_;           // 1,048,576
    size_t i = (size_t)blockIdx.x * blockDim.x + threadIdx.x;
    if (i < NX) {
        xb[i] = f2bf(x[i]);
    } else {
        size_t j = i - NX;                       // j < 4*NW
        size_t sel = j >> 20;                    // NW == 1<<20
        size_t off = j & (NW - 1);
        const float* src = (sel == 0) ? wq : (sel == 1) ? wk : (sel == 2) ? wv : wp;
        wb[j] = f2bf(src[off]);
    }
}

// ---------------------------------------------------------------------------
// Per-wave GEMM core: 32x128 output tile (2 M-subtiles x 8 N-frags),
// K-loop in steps of 32.  16 WMMAs per 20 b128 loads.
// A : [M x K] bf16 row-major (rows m0..m0+31)
// W : [N x K] bf16 row-major (rows n0..n0+127) -> out = A * W^T
// ---------------------------------------------------------------------------
__device__ __forceinline__ void gemm_tile_32x128(
    const u16* __restrict__ A, const u16* __restrict__ W,
    int Kdim, int m0, int n0, int lane, v8f acc[2][8])
{
    const int half = lane >> 4;       // 0: lanes 0-15, 1: lanes 16-31
    const int lm   = lane & 15;
    const u16* arow0 = A + (size_t)(m0 + lm) * Kdim + (half ? 8 : 0);
    const u16* arow1 = arow0 + (size_t)16 * Kdim;
    for (int k0 = 0; k0 < Kdim; k0 += 32) {
        Frag a0, a1;
        a0.q[0] = *(const uint4*)(arow0 + k0);
        a0.q[1] = *(const uint4*)(arow0 + k0 + 16);
        a1.q[0] = *(const uint4*)(arow1 + k0);
        a1.q[1] = *(const uint4*)(arow1 + k0 + 16);
        if (k0 + 32 < Kdim) {   // stream next A cachelines (global_prefetch_b8)
            __builtin_prefetch((const void*)(arow0 + k0 + 32), 0, 1);
            __builtin_prefetch((const void*)(arow1 + k0 + 32), 0, 1);
        }
#pragma unroll
        for (int nf = 0; nf < 8; ++nf) {
            Frag b;
            const u16* wrow = W + (size_t)(n0 + nf * 16 + lm) * Kdim + k0 + (half ? 8 : 0);
            b.q[0] = *(const uint4*)(wrow);
            b.q[1] = *(const uint4*)(wrow + 16);
            acc[0][nf] = WMMA_BF16(a0.v, b.v, acc[0][nf]);
            acc[1][nf] = WMMA_BF16(a1.v, b.v, acc[1][nf]);
        }
    }
}

// ---------------------------------------------------------------------------
// Kernel 1: QKV projection.  blockIdx.z selects Wq/Wk/Wv.
// Q,K stored bf16 [B,H,T,D]; V stored transposed bf16 [B,H,D,T].
// ---------------------------------------------------------------------------
__global__ __launch_bounds__(32) void qkv_gemm_kernel(
    const u16* __restrict__ xb, const u16* __restrict__ wb,
    u16* __restrict__ Q, u16* __restrict__ Kp, u16* __restrict__ Vt)
{
    const int lane = threadIdx.x;
    const int n0 = blockIdx.x * 128;
    const int m0 = blockIdx.y * 32;
    const int mode = blockIdx.z;          // 0=q, 1=k, 2=v
    const u16* W = wb + (size_t)mode * C_ * C_;

    v8f acc[2][8] = {};
    gemm_tile_32x128(xb, W, C_, m0, n0, lane, acc);

    const int half = lane >> 4;
    const int lm = lane & 15;
    const int b = m0 / T_;
    const int t0 = m0 - b * T_;
#pragma unroll
    for (int nf = 0; nf < 8; ++nf) {
        const int n = n0 + nf * 16 + lm;
        const int h = n >> 6;
        const int d = n & 63;
#pragma unroll
        for (int mt = 0; mt < 2; ++mt) {
#pragma unroll
            for (int r = 0; r < 8; ++r) {
                const int t = t0 + mt * 16 + r + half * 8;
                const u16 val = f2bf(acc[mt][nf][r]);
                if (mode == 2) {
                    Vt[(((size_t)b * H_ + h) * D_ + d) * T_ + t] = val;
                } else {
                    u16* dst = (mode == 0) ? Q : Kp;
                    dst[(((size_t)b * H_ + h) * T_ + t) * D_ + d] = val;
                }
            }
        }
    }
}

// ---------------------------------------------------------------------------
// Kernel 2: causal flash attention, one wave per (b, h, 32-query tile).
// Q,K: bf16 [B,H,T,D]; Vt: bf16 [B,H,D,T]; Y: bf16 [B,T,C] (= [B,T,H*D]).
// ---------------------------------------------------------------------------
__global__ __launch_bounds__(32) void attn_kernel(
    const u16* __restrict__ Q, const u16* __restrict__ Kp,
    const u16* __restrict__ Vt, u16* __restrict__ Y)
{
    __shared__ __align__(16) u16 Plds[32 * 32];   // P tile: 32 q-rows x 32 keys

    const int lane = threadIdx.x;
    const int half = lane >> 4;
    const int lm = lane & 15;
    const int qt = blockIdx.x;
    const int h  = blockIdx.y;
    const int b  = blockIdx.z;
    const size_t bh = (size_t)b * H_ + h;
    const int m0 = qt * 32;

    // Q A-fragments: 2 M-subtiles x (d=0..31, d=32..63)
    Frag aQ[2][2];
#pragma unroll
    for (int mt = 0; mt < 2; ++mt) {
        const u16* qrow = Q + (bh * T_ + m0 + mt * 16 + lm) * D_ + (half ? 8 : 0);
        aQ[mt][0].q[0] = *(const uint4*)(qrow);
        aQ[mt][0].q[1] = *(const uint4*)(qrow + 16);
        aQ[mt][1].q[0] = *(const uint4*)(qrow + 32);
        aQ[mt][1].q[1] = *(const uint4*)(qrow + 48);
    }

    v8f accO[2][4] = {};
    float mrow[2][8], lrow[2][8];
#pragma unroll
    for (int mt = 0; mt < 2; ++mt)
#pragma unroll
        for (int r = 0; r < 8; ++r) { mrow[mt][r] = NEG_BIG; lrow[mt][r] = 0.0f; }

    const int nb = (m0 + 31) / 32 + 1;    // 32-key blocks needed (causal)
    for (int kbi = 0; kbi < nb; ++kbi) {
        const int kb = kbi * 32;

        // ---- S = Q K^T : K-frags shared across both M-subtiles ----
        v8f s[2][2];
#pragma unroll
        for (int sub = 0; sub < 2; ++sub) {
            const int kcol = kb + sub * 16 + lm;
            const u16* krow = Kp + (bh * T_ + kcol) * D_ + (half ? 8 : 0);
            Frag b0, b1;
            b0.q[0] = *(const uint4*)(krow);
            b0.q[1] = *(const uint4*)(krow + 16);
            b1.q[0] = *(const uint4*)(krow + 32);
            b1.q[1] = *(const uint4*)(krow + 48);
#pragma unroll
            for (int mt = 0; mt < 2; ++mt) {
                v8f z = {};
                z = WMMA_BF16(aQ[mt][0].v, b0.v, z);
                z = WMMA_BF16(aQ[mt][1].v, b1.v, z);
                s[mt][sub] = z;
            }
        }

        // ---- scale + causal mask ----
#pragma unroll
        for (int mt = 0; mt < 2; ++mt)
#pragma unroll
            for (int r = 0; r < 8; ++r) {
                const int row = m0 + mt * 16 + r + half * 8;
#pragma unroll
                for (int sub = 0; sub < 2; ++sub) {
                    const int col = kb + sub * 16 + lm;
                    const float v = s[mt][sub][r] * 0.125f;   // 1/sqrt(64)
                    s[mt][sub][r] = (col <= row) ? v : NEG_BIG;
                }
            }

        // ---- online softmax (16-lane halves own rows r / r+8) ----
        float fac[2][8];
#pragma unroll
        for (int mt = 0; mt < 2; ++mt)
#pragma unroll
            for (int r = 0; r < 8; ++r) {
                float mx = fmaxf(s[mt][0][r], s[mt][1][r]);
#pragma unroll
                for (int off = 8; off >= 1; off >>= 1)
                    mx = fmaxf(mx, __shfl_xor(mx, off, 32));
                const float mnew = fmaxf(mrow[mt][r], mx);
                const float f = exp2f((mrow[mt][r] - mnew) * LOG2E);
                const float p0 = exp2f((s[mt][0][r] - mnew) * LOG2E);
                const float p1 = exp2f((s[mt][1][r] - mnew) * LOG2E);
                s[mt][0][r] = p0; s[mt][1][r] = p1;
                float ps = p0 + p1;
#pragma unroll
                for (int off = 8; off >= 1; off >>= 1)
                    ps += __shfl_xor(ps, off, 32);
                lrow[mt][r] = lrow[mt][r] * f + ps;
                mrow[mt][r] = mnew;
                fac[mt][r] = f;
            }
#pragma unroll
        for (int mt = 0; mt < 2; ++mt)
#pragma unroll
            for (int nf = 0; nf < 4; ++nf)
#pragma unroll
                for (int r = 0; r < 8; ++r)
                    accO[mt][nf][r] *= fac[mt][r];

        // ---- stage P (32x32 bf16) through LDS: C-layout -> A-layout ----
        __syncthreads();
#pragma unroll
        for (int mt = 0; mt < 2; ++mt)
#pragma unroll
            for (int r = 0; r < 8; ++r) {
                const int row = mt * 16 + r + half * 8;
                Plds[row * 32 + lm]      = f2bf(s[mt][0][r]);
                Plds[row * 32 + 16 + lm] = f2bf(s[mt][1][r]);
            }
        __syncthreads();
        Frag aP[2];
#pragma unroll
        for (int mt = 0; mt < 2; ++mt) {
            const u16* prow = &Plds[(mt * 16 + lm) * 32 + (half ? 8 : 0)];
            aP[mt].q[0] = *(const uint4*)(prow);
            aP[mt].q[1] = *(const uint4*)(prow + 16);
        }

        // ---- O += P * V  (V-frags shared across both M-subtiles) ----
#pragma unroll
        for (int nf = 0; nf < 4; ++nf) {
            const int d = nf * 16 + lm;
            const u16* vrow = Vt + (bh * D_ + d) * T_ + kb + (half ? 8 : 0);
            Frag bv;
            bv.q[0] = *(const uint4*)(vrow);
            bv.q[1] = *(const uint4*)(vrow + 16);
#pragma unroll
            for (int mt = 0; mt < 2; ++mt)
                accO[mt][nf] = WMMA_BF16(aP[mt].v, bv.v, accO[mt][nf]);
        }
    }

    // ---- epilogue: O /= l, write Y bf16 [B,T,C] ----
#pragma unroll
    for (int mt = 0; mt < 2; ++mt) {
        float invl[8];
#pragma unroll
        for (int r = 0; r < 8; ++r) invl[r] = 1.0f / lrow[mt][r];
#pragma unroll
        for (int nf = 0; nf < 4; ++nf) {
            const int c = h * D_ + nf * 16 + lm;
#pragma unroll
            for (int r = 0; r < 8; ++r) {
                const int t = m0 + mt * 16 + r + half * 8;
                Y[((size_t)b * T_ + t) * C_ + c] = f2bf(accO[mt][nf][r] * invl[r]);
            }
        }
    }
}

// ---------------------------------------------------------------------------
// Kernel 3: output projection, fp32 result to d_out.
// ---------------------------------------------------------------------------
__global__ __launch_bounds__(32) void proj_gemm_kernel(
    const u16* __restrict__ Y, const u16* __restrict__ Wp,
    float* __restrict__ out)
{
    const int lane = threadIdx.x;
    const int n0 = blockIdx.x * 128;
    const int m0 = blockIdx.y * 32;

    v8f acc[2][8] = {};
    gemm_tile_32x128(Y, Wp, C_, m0, n0, lane, acc);

    const int half = lane >> 4;
    const int lm = lane & 15;
#pragma unroll
    for (int nf = 0; nf < 8; ++nf) {
        const int n = n0 + nf * 16 + lm;
#pragma unroll
        for (int mt = 0; mt < 2; ++mt) {
#pragma unroll
            for (int r = 0; r < 8; ++r) {
                const int m = m0 + mt * 16 + r + half * 8;
                out[(size_t)m * C_ + n] = acc[mt][nf][r];
            }
        }
    }
}

// ---------------------------------------------------------------------------
// Launcher
// ---------------------------------------------------------------------------
extern "C" void kernel_launch(void* const* d_in, const int* in_sizes, int n_in,
                              void* d_out, int out_size, void* d_ws, size_t ws_size,
                              hipStream_t stream) {
    const float* x  = (const float*)d_in[0];
    const float* wq = (const float*)d_in[1];
    const float* wk = (const float*)d_in[2];
    const float* wv = (const float*)d_in[3];
    const float* wp = (const float*)d_in[4];

    const size_t NX = (size_t)B_ * T_ * C_;   // 4,194,304 elems
    const size_t NW = (size_t)C_ * C_;        // 1,048,576 elems

    u16* xb = (u16*)d_ws;            // bf16 x              : 8 MiB
    u16* wb = xb + NX;               // bf16 Wq|Wk|Wv|Wproj : 8 MiB
    u16* Qb = wb + 4 * NW;           // bf16 Q [B,H,T,D]    : 8 MiB
    u16* Kb = Qb + NX;               // bf16 K [B,H,T,D]    : 8 MiB
    u16* Vt = Kb + NX;               // bf16 V^T [B,H,D,T]  : 8 MiB
    u16* Yb = Vt + NX;               // bf16 Y [B,T,C]      : 8 MiB

    // 0) convert inputs to bf16
    {
        const size_t total = NX + 4 * NW;     // 8,388,608
        cvt_bf16_kernel<<<dim3((unsigned)(total / 256)), dim3(256), 0, stream>>>(
            x, wq, wk, wv, wp, xb, wb);
    }
    // 1) QKV projections (V transposed)
    {
        dim3 grid(C_ / 128, (B_ * T_) / 32, 3);   // (8, 128, 3)
        qkv_gemm_kernel<<<grid, dim3(32), 0, stream>>>(xb, wb, Qb, Kb, Vt);
    }
    // 2) causal flash attention
    {
        dim3 grid(T_ / 32, H_, B_);               // (64, 16, 2)
        attn_kernel<<<grid, dim3(32), 0, stream>>>(Qb, Kb, Vt, Yb);
    }
    // 3) output projection -> fp32 d_out
    {
        dim3 grid(C_ / 128, (B_ * T_) / 32, 1);   // (8, 128)
        proj_gemm_kernel<<<grid, dim3(32), 0, stream>>>(Yb, wb + 3 * NW, (float*)d_out);
    }
}
